// MessagePassingLayer_84774064488460
// MI455X (gfx1250) — compile-verified, matrix operand order checked by reading, not run on hardware
//
#include <hip/hip_runtime.h>
#include <hip/hip_bf16.h>

// ---------------------------------------------------------------------------
// MessagePassingLayer for MI455X (gfx1250, wave32, WMMA)
//
// GEMMs run on v_wmma_f32_16x16x32_bf16 (bf16 in, f32 accumulate).
// Edge softmax/scatter phase uses L2-resident float atomics.
// ---------------------------------------------------------------------------

typedef __attribute__((ext_vector_type(16))) __bf16 v16bf;
typedef __attribute__((ext_vector_type(8)))  float  v8f;

#define GF_ACCUM 1
#define GF_BIAS  2
#define GF_GELU  4

#define TM 128   // rows per block (8 waves x 16)
#define TN 64    // cols per block (4 wmma tiles per wave)
#define TK 64    // k per staging step (2 wmma k-substeps of 32)
#define LDS_PAD 8

__device__ __forceinline__ float gelu_exact(float x) {
    return 0.5f * x * (1.0f + erff(x * 0.70710678118654752f));
}

__device__ __forceinline__ uint2 pack4_bf16(float4 v) {
    union { __bf16 h[4]; uint2 u; } p;
    p.h[0] = (__bf16)v.x; p.h[1] = (__bf16)v.y;
    p.h[2] = (__bf16)v.z; p.h[3] = (__bf16)v.w;
    return p.u;
}

// C[M,N] (+)= gelu(A[M,K] @ B[K,N] + bias), all row-major f32 in memory,
// bf16 on the matrix pipes.
__global__ __launch_bounds__(256)
void gemm_bf16_wmma(const float* __restrict__ A, const float* __restrict__ B,
                    const float* __restrict__ bias, float* __restrict__ C,
                    int M, int N, int K, int flags)
{
    __shared__ __bf16 As[TM][TK + LDS_PAD];   // [m][k]
    __shared__ __bf16 Bs[TN][TK + LDS_PAD];   // [n][k]  (transposed tile)

    const int tid  = threadIdx.x;
    const int wave = tid >> 5;          // 0..7
    const int lane = tid & 31;
    const int lrow = lane & 15;
    const int half = lane >> 4;         // 0/1
    const int m0   = blockIdx.x * TM;
    const int n0   = blockIdx.y * TN;

    v8f acc[4];
#pragma unroll
    for (int t = 0; t < 4; ++t)
#pragma unroll
        for (int j = 0; j < 8; ++j) acc[t][j] = 0.0f;

    for (int k0 = 0; k0 < K; k0 += TK) {
        // ---- stage A tile: 128x64 elems = 2048 float4, 8 per thread ------
        // Phase 1: issue all global loads (rows clamped in-bounds so loads
        // stay unconditional and batch); Phase 2: convert + LDS store.
        float4 av[8];
#pragma unroll
        for (int i = 0; i < 8; ++i) {
            int linear = i * 256 + tid;      // 0..2047
            int r  = linear >> 4;            // 16 float4 per row
            int c4 = linear & 15;
            int m  = m0 + r;
            int mc = (m < M) ? m : (M - 1);
            av[i] = *(const float4*)(A + (size_t)mc * K + k0 + c4 * 4);
        }
        float4 bv[4];
#pragma unroll
        for (int i = 0; i < 4; ++i) {
            int linear = i * 256 + tid;
            int kk = linear >> 4;
            int c4 = linear & 15;
            bv[i] = *(const float4*)(B + (size_t)(k0 + kk) * N + n0 + c4 * 4);
        }
#pragma unroll
        for (int i = 0; i < 8; ++i) {
            int linear = i * 256 + tid;
            int r  = linear >> 4;
            int c4 = linear & 15;
            int m  = m0 + r;
            float4 v = av[i];
            if (m >= M) { v.x = 0.f; v.y = 0.f; v.z = 0.f; v.w = 0.f; }
            *(uint2*)&As[r][c4 * 4] = pack4_bf16(v);
        }
#pragma unroll
        for (int i = 0; i < 4; ++i) {
            int linear = i * 256 + tid;
            int kk = linear >> 4;
            int c4 = linear & 15;
            float4 v = bv[i];
            Bs[c4 * 4 + 0][kk] = (__bf16)v.x;
            Bs[c4 * 4 + 1][kk] = (__bf16)v.y;
            Bs[c4 * 4 + 2][kk] = (__bf16)v.z;
            Bs[c4 * 4 + 3][kk] = (__bf16)v.w;
        }
        __syncthreads();

        // ---- compute: preload ALL fragments (20x ds_load_b128), then run
        // the 8 WMMAs back-to-back without per-op LDS stalls --------------
        {
            const int r = wave * 16 + lrow;
            v16bf afr[2];
            v16bf bfr[2][4];
#pragma unroll
            for (int ks2 = 0; ks2 < 2; ++ks2) {
                const int ks = ks2 * 32;
#pragma unroll
                for (int i = 0; i < 8; ++i) {
                    afr[ks2][i]     = As[r][ks + half * 8 + i];
                    afr[ks2][8 + i] = As[r][ks + 16 + half * 8 + i];
                }
#pragma unroll
                for (int t = 0; t < 4; ++t) {
                    const int c = t * 16 + lrow;
#pragma unroll
                    for (int i = 0; i < 8; ++i) {
                        bfr[ks2][t][i]     = Bs[c][ks + half * 8 + i];
                        bfr[ks2][t][8 + i] = Bs[c][ks + 16 + half * 8 + i];
                    }
                }
            }
#pragma unroll
            for (int ks2 = 0; ks2 < 2; ++ks2)
#pragma unroll
                for (int t = 0; t < 4; ++t)
                    acc[t] = __builtin_amdgcn_wmma_f32_16x16x32_bf16(
                        false, afr[ks2], false, bfr[ks2][t], (short)0,
                        acc[t], false, false);
        }
        __syncthreads();
    }

    // epilogue: D vgpr j -> row = half*8 + j, col = lrow  (ISA C/D layout)
#pragma unroll
    for (int t = 0; t < 4; ++t) {
        int n = n0 + t * 16 + lrow;
#pragma unroll
        for (int j = 0; j < 8; ++j) {
            int m = m0 + wave * 16 + half * 8 + j;
            if (m < M) {
                size_t off = (size_t)m * N + n;
                float v = acc[t][j];
                if (flags & GF_ACCUM) v += C[off];
                if (flags & GF_BIAS)  v += bias[n];
                if (flags & GF_GELU)  v = gelu_exact(v);
                C[off] = v;
            }
        }
    }
}

// ---------------------------------------------------------------------------
// Edge attention kernels (H=4 heads, DK=64)
// ---------------------------------------------------------------------------
#define NH 4
#define DKH 64

// score[e,h] = <Q[tgt],K[src]>_h / 8 + log(clip(nw)); atomic segment max vs 0
__global__ void edge_scores_kernel(const float* __restrict__ Q,
                                   const float* __restrict__ Kf,
                                   const int* __restrict__ tgt,
                                   const int* __restrict__ src,
                                   const float* __restrict__ nw,
                                   float* __restrict__ scores,
                                   float* __restrict__ smax, int E)
{
    int idx = blockIdx.x * blockDim.x + threadIdx.x;
    if (idx >= E * NH) return;
    int e = idx >> 2, h = idx & 3;
    int t = tgt[e], s = src[e];
    const float4* q = (const float4*)(Q  + (size_t)t * 256 + h * DKH);
    const float4* k = (const float4*)(Kf + (size_t)s * 256 + h * DKH);
    float acc = 0.0f;
#pragma unroll
    for (int i = 0; i < 16; ++i) {
        float4 a = q[i], b = k[i];
        acc += a.x * b.x + a.y * b.y + a.z * b.z + a.w * b.w;
    }
    float sc = acc * 0.125f + logf(fmaxf(nw[e], 1e-10f));
    scores[(size_t)e * NH + h] = sc;
    // smax initialized to 0.0 (reference max includes 0); only positive
    // scores can raise it, and for positive IEEE floats the int-bit order
    // matches the float order, so integer atomicMax is exact.
    if (sc > 0.0f)
        atomicMax((int*)&smax[(size_t)t * NH + h], __float_as_int(sc));
}

__global__ void edge_expsum_kernel(float* __restrict__ scores,
                                   const float* __restrict__ smax,
                                   const int* __restrict__ tgt,
                                   float* __restrict__ sumex, int E)
{
    int idx = blockIdx.x * blockDim.x + threadIdx.x;
    if (idx >= E * NH) return;
    int e = idx >> 2, h = idx & 3;
    int t = tgt[e];
    float ex = expf(scores[idx] - smax[(size_t)t * NH + h]);
    scores[idx] = ex;
    atomicAdd(&sumex[(size_t)t * NH + h], ex);
}

// msg[tgt] += (ex / (sum+1e-10)) * V[src]; one thread per (edge, 4 dims)
__global__ void edge_scatter_kernel(const float* __restrict__ scores,
                                    const float* __restrict__ sumex,
                                    const float* __restrict__ V,
                                    const int* __restrict__ tgt,
                                    const int* __restrict__ src,
                                    float* __restrict__ msg, int E)
{
    int idx = blockIdx.x * blockDim.x + threadIdx.x;
    if (idx >= E * 64) return;
    int e  = idx >> 6;
    int d0 = (idx & 63) * 4;
    int h  = d0 >> 6;
    int t = tgt[e], s = src[e];
    float ex  = scores[(size_t)e * NH + h];
    float att = ex / (sumex[(size_t)t * NH + h] + 1e-10f);
    float4 v = *(const float4*)(V + (size_t)s * 256 + d0);
    float* o = msg + (size_t)t * 256 + d0;
    atomicAdd(o + 0, att * v.x);
    atomicAdd(o + 1, att * v.y);
    atomicAdd(o + 2, att * v.z);
    atomicAdd(o + 3, att * v.w);
}

// ---------------------------------------------------------------------------
// Host orchestration
// ---------------------------------------------------------------------------
static float* carve(char*& p, size_t nfloats) {
    float* r = (float*)p;
    size_t bytes = (nfloats * sizeof(float) + 255) & ~(size_t)255;
    p += bytes;
    return r;
}

static void gemm(const float* A, const float* B, const float* bias, float* C,
                 int M, int N, int K, int flags, hipStream_t stream) {
    dim3 grid((M + TM - 1) / TM, N / TN);
    gemm_bf16_wmma<<<grid, 256, 0, stream>>>(A, B, bias, C, M, N, K, flags);
}

extern "C" void kernel_launch(void* const* d_in, const int* in_sizes, int n_in,
                              void* d_out, int out_size, void* d_ws, size_t ws_size,
                              hipStream_t stream) {
    const float* inv_h   = (const float*)d_in[0];
    const float* asset_h = (const float*)d_in[1];
    const float* inv_nw  = (const float*)d_in[2];
    const float* ast_nw  = (const float*)d_in[3];
    const float* m_w1    = (const float*)d_in[4];
    const float* m_b1    = (const float*)d_in[5];
    const float* m_w2    = (const float*)d_in[6];
    const float* m_b2    = (const float*)d_in[7];
    const float* Wq      = (const float*)d_in[8];
    const float* Wk      = (const float*)d_in[9];
    const float* Wv      = (const float*)d_in[10];
    const float* u_w1    = (const float*)d_in[11];
    const float* u_b1    = (const float*)d_in[12];
    const float* u_w2    = (const float*)d_in[13];
    const float* u_b2    = (const float*)d_in[14];
    const int*   e_tgt   = (const int*)d_in[15];
    const int*   e_src   = (const int*)d_in[16];

    const int D = 256;
    const int I = in_sizes[0] / D;
    const int A = in_sizes[1] / D;
    const int E = in_sizes[15];

    char* p = (char*)d_ws;
    float* tmpI    = carve(p, (size_t)I * D);
    float* tmpA    = carve(p, (size_t)A * D);
    float* M_inv   = carve(p, (size_t)I * D);
    float* M_ast   = carve(p, (size_t)A * D);
    float* Q_inv   = carve(p, (size_t)I * D);
    float* K_inv   = carve(p, (size_t)I * D);
    float* V_inv   = carve(p, (size_t)I * D);
    float* Q_ast   = carve(p, (size_t)A * D);
    float* K_ast   = carve(p, (size_t)A * D);
    float* V_ast   = carve(p, (size_t)A * D);
    float* inv_msg = carve(p, (size_t)I * D);
    float* ast_msg = carve(p, (size_t)A * D);
    float* scores  = carve(p, (size_t)E * NH);
    float* smaxI   = carve(p, (size_t)I * NH);
    float* sumI    = carve(p, (size_t)I * NH);
    float* smaxA   = carve(p, (size_t)A * NH);
    float* sumA    = carve(p, (size_t)A * NH);

    float* out_inv = (float*)d_out;
    float* out_ast = out_inv + (size_t)I * D;

    // -------- message MLPs: M = gelu(gelu(h @ m_w1 + b1) @ m_w2 + b2) ------
    gemm(asset_h, m_w1, m_b1, tmpA,  A, D, D, GF_BIAS | GF_GELU, stream);
    gemm(tmpA,    m_w2, m_b2, M_ast, A, D, D, GF_BIAS | GF_GELU, stream);
    gemm(inv_h,   m_w1, m_b1, tmpI,  I, D, D, GF_BIAS | GF_GELU, stream);
    gemm(tmpI,    m_w2, m_b2, M_inv, I, D, D, GF_BIAS | GF_GELU, stream);

    // -------- Q/K/V projections -------------------------------------------
    gemm(inv_h,   Wq, nullptr, Q_inv, I, D, D, 0, stream);
    gemm(inv_h,   Wk, nullptr, K_inv, I, D, D, 0, stream);
    gemm(M_inv,   Wv, nullptr, V_inv, I, D, D, 0, stream);
    gemm(asset_h, Wq, nullptr, Q_ast, A, D, D, 0, stream);
    gemm(asset_h, Wk, nullptr, K_ast, A, D, D, 0, stream);
    gemm(M_ast,   Wv, nullptr, V_ast, A, D, D, 0, stream);

    // -------- zero-init segment state (graph-capture-safe memsets) --------
    hipMemsetAsync(smaxI,   0, (size_t)I * NH * sizeof(float), stream);
    hipMemsetAsync(sumI,    0, (size_t)I * NH * sizeof(float), stream);
    hipMemsetAsync(smaxA,   0, (size_t)A * NH * sizeof(float), stream);
    hipMemsetAsync(sumA,    0, (size_t)A * NH * sizeof(float), stream);
    hipMemsetAsync(inv_msg, 0, (size_t)I * D  * sizeof(float), stream);
    hipMemsetAsync(ast_msg, 0, (size_t)A * D  * sizeof(float), stream);

    const int TB = 256;
    int gEH = (E * NH + TB - 1) / TB;
    int gE64 = (E * 64 + TB - 1) / TB;

    // -------- direction 1: investors <- assets ----------------------------
    edge_scores_kernel<<<gEH, TB, 0, stream>>>(Q_inv, K_ast, e_tgt, e_src,
                                               inv_nw, scores, smaxI, E);
    edge_expsum_kernel<<<gEH, TB, 0, stream>>>(scores, smaxI, e_tgt, sumI, E);
    edge_scatter_kernel<<<gE64, TB, 0, stream>>>(scores, sumI, V_ast,
                                                 e_tgt, e_src, inv_msg, E);

    // -------- direction 2: assets <- investors (reversed edges) -----------
    edge_scores_kernel<<<gEH, TB, 0, stream>>>(Q_ast, K_inv, e_src, e_tgt,
                                               ast_nw, scores, smaxA, E);
    edge_expsum_kernel<<<gEH, TB, 0, stream>>>(scores, smaxA, e_src, sumA, E);
    edge_scatter_kernel<<<gE64, TB, 0, stream>>>(scores, sumA, V_inv,
                                                 e_src, e_tgt, ast_msg, E);

    // -------- update MLP: gelu(gelu([h, msg] @ u_w1 + b1) @ u_w2 + b2) ----
    // concat GEMM split into two K=256 passes (second pass accumulates)
    const float* u_w1_bot = u_w1 + (size_t)D * D;  // rows 256..511
    gemm(inv_h,   u_w1,     nullptr, tmpI, I, D, D, 0, stream);
    gemm(inv_msg, u_w1_bot, u_b1,    tmpI, I, D, D, GF_ACCUM | GF_BIAS | GF_GELU, stream);
    gemm(tmpI,    u_w2,     u_b2,    out_inv, I, D, D, GF_BIAS | GF_GELU, stream);

    gemm(asset_h, u_w1,     nullptr, tmpA, A, D, D, 0, stream);
    gemm(ast_msg, u_w1_bot, u_b1,    tmpA, A, D, D, GF_ACCUM | GF_BIAS | GF_GELU, stream);
    gemm(tmpA,    u_w2,     u_b2,    out_ast, A, D, D, GF_BIAS | GF_GELU, stream);

    (void)n_in; (void)out_size; (void)ws_size;
}
